// OSL2_35837207118338
// MI455X (gfx1250) — matrix operand order repaired and checked by reading, not compiled, not run on hardware
//
#include <hip/hip_runtime.h>

typedef __attribute__((ext_vector_type(16))) _Float16 v16h;
typedef __attribute__((ext_vector_type(8)))  float    v8f;

#define WAVE 32

__device__ __forceinline__ float wave_sum_f32(float v) {
#pragma unroll
  for (int off = 16; off >= 1; off >>= 1) v += __shfl_xor(v, off, WAVE);
  return v;
}

__device__ __forceinline__ void online_upd(float v, float& m, float& ss) {
  if (v > m) { ss = ss * __expf(m - v) + 1.0f; m = v; }
  else       { ss += __expf(v - m); }
}

__device__ __forceinline__ void online_upd_idx(float v, int idx, float& m,
                                               float& ss, int& am) {
  if (v > m) { ss = ss * __expf(m - v) + 1.0f; m = v; am = idx; }
  else       { ss += __expf(v - m); }
}

union F8  { float4 q[2]; float f[8];  };
union F16 { float4 q[4]; float f[16]; };

// ---------------------------------------------------------------------------
// Kernel 0: zero accumulator [C,C], count [C], and the scalar output.
// ---------------------------------------------------------------------------
__global__ void osl_zero(float* __restrict__ acc, float* __restrict__ count,
                         float* __restrict__ out, int C) {
  long long i = ((long long)blockIdx.x * blockDim.x + threadIdx.x) * 4;
  long long cc = (long long)C * C;
  long long n  = cc + C;
  if (i == 0) out[0] = 0.0f;
  if (i + 3 < cc) {
    *(float4*)(acc + i) = make_float4(0.f, 0.f, 0.f, 0.f);
  } else {
    for (long long k = i; k < i + 4 && k < n; ++k) {
      if (k < cc) acc[k] = 0.0f;
      else        count[k - cc] = 0.0f;
    }
  }
}

// ---------------------------------------------------------------------------
// Kernel 1: one wave per row of logits_clean. Online softmax + argmax in a
// single vectorized read; if clean-correct, scatter probs into acc[target]
// (L2-resident f32 atomics) and bump count[target].
// ---------------------------------------------------------------------------
__global__ void osl_stats(const float* __restrict__ clean,
                          const int* __restrict__ target,
                          float* __restrict__ acc, float* __restrict__ count,
                          int B, int C) {
  const int lane = threadIdx.x & (WAVE - 1);
  const int row  = (blockIdx.x * blockDim.x + threadIdx.x) / WAVE;
  if (row >= B) return;

  const float*  x  = clean + (long long)row * C;
  const float4* x4 = (const float4*)x;
  const int nv = C >> 2;

  float m = -3.4e38f, s = 0.0f;
  int   am = 0x7fffffff;
  for (int j = lane; j < nv; j += WAVE) {
    float4 v = x4[j];
    int c0 = j << 2;
    online_upd_idx(v.x, c0 + 0, m, s, am);
    online_upd_idx(v.y, c0 + 1, m, s, am);
    online_upd_idx(v.z, c0 + 2, m, s, am);
    online_upd_idx(v.w, c0 + 3, m, s, am);
  }
  for (int j = (nv << 2) + lane; j < C; j += WAVE)   // scalar tail (C%4)
    online_upd_idx(x[j], j, m, s, am);

  // butterfly merge of (max, sumexp, argmax[first-hit tie-break])
#pragma unroll
  for (int off = 16; off >= 1; off >>= 1) {
    float om = __shfl_xor(m, off, WAVE);
    float os = __shfl_xor(s, off, WAVE);
    int   oa = __shfl_xor(am, off, WAVE);
    float nm = fmaxf(m, om);
    s = s * __expf(m - nm) + os * __expf(om - nm);
    if (om > m || (om == m && oa < am)) am = oa;
    m = nm;
  }

  const int tgt = target[row];
  if (am == tgt) {                 // clean-correct: contribute probs
    float inv = 1.0f / s;
    float* arow = acc + (long long)tgt * C;
    for (int j = lane; j < nv; j += WAVE) {
      float4 v = x4[j];
      int c0 = j << 2;
      atomicAdd(&arow[c0 + 0], __expf(v.x - m) * inv);
      atomicAdd(&arow[c0 + 1], __expf(v.y - m) * inv);
      atomicAdd(&arow[c0 + 2], __expf(v.z - m) * inv);
      atomicAdd(&arow[c0 + 3], __expf(v.w - m) * inv);
    }
    for (int j = (nv << 2) + lane; j < C; j += WAVE)
      atomicAdd(&arow[j], __expf(x[j] - m) * inv);
    if (lane == 0) atomicAdd(&count[tgt], 1.0f);
  }
}

// ---------------------------------------------------------------------------
// Kernel 2: one wave per class row. matrix = acc / rowsum (count cancels
// algebraically); unseen classes get a one-hot row.
// ---------------------------------------------------------------------------
__global__ void osl_matrix(const float* __restrict__ acc,
                           const float* __restrict__ count,
                           float* __restrict__ matrix, int C) {
  const int lane = threadIdx.x & (WAVE - 1);
  const int c    = (blockIdx.x * blockDim.x + threadIdx.x) / WAVE;
  if (c >= C) return;
  const float*  arow = acc + (long long)c * C;
  const float4* a4   = (const float4*)arow;
  const int nv = C >> 2;

  float ps = 0.0f;
  for (int j = lane; j < nv; j += WAVE) {
    float4 v = a4[j];
    ps += v.x + v.y + v.z + v.w;
  }
  for (int j = (nv << 2) + lane; j < C; j += WAVE) ps += arow[j];
  float srow = wave_sum_f32(ps);

  float cnt = count[c];
  float* mrow = matrix + (long long)c * C;
  if (cnt == 0.0f) {
    for (int j = lane; j < C; j += WAVE) mrow[j] = (j == c) ? 1.0f : 0.0f;
  } else {
    float inv = (srow > 0.0f) ? (1.0f / srow) : 0.0f;
    for (int j = lane; j < nv; j += WAVE) {
      float4 v = a4[j];
      *(float4*)(mrow + (j << 2)) =
          make_float4(v.x * inv, v.y * inv, v.z * inv, v.w * inv);
    }
    for (int j = (nv << 2) + lane; j < C; j += WAVE) mrow[j] = arow[j] * inv;
  }
}

// ---------------------------------------------------------------------------
// Kernel 3: one wave per 16 rows of logits_adv.
// Phase A: cooperative online logZ per row (vectorized, coalesced read).
// Phase B: WMMA contraction. A = softlabel tile (16x32 f16 from L2-resident
//          matrix), B = (logZ - x) tile (32x16 f16). diag(D) = per-row loss.
// Full 32-wide K chunks: unguarded b128 loads. Tail chunk: clamped addresses
// + register selects (branch-free, EXEC stays all-ones).
// ---------------------------------------------------------------------------
__global__ void osl_loss(const float* __restrict__ adv,
                         const int* __restrict__ target,
                         const float* __restrict__ matrix,
                         float* __restrict__ out, int B, int C) {
  const int lane    = threadIdx.x & (WAVE - 1);
  const int wid     = (blockIdx.x * blockDim.x + threadIdx.x) / WAVE;
  const int rowBase = wid * 16;
  if (rowBase >= B) return;
  const int s  = lane & 15;        // sample this lane serves (A row / B col)
  const int hi = (lane >> 4) & 1;  // lane half (K-split: k = 16*hi + element)

  // ---- Phase A: logZ for each of the 16 rows ----
  float own_logZ = 0.0f;
  const int nv = C >> 2;
  for (int r = 0; r < 16; ++r) {
    const float*  x  = adv + (long long)(rowBase + r) * C;
    const float4* x4 = (const float4*)x;
    float m = -3.4e38f, ss = 0.0f;
    for (int j = lane; j < nv; j += WAVE) {
      float4 v = x4[j];
      online_upd(v.x, m, ss);
      online_upd(v.y, m, ss);
      online_upd(v.z, m, ss);
      online_upd(v.w, m, ss);
    }
    for (int j = (nv << 2) + lane; j < C; j += WAVE) online_upd(x[j], m, ss);
#pragma unroll
    for (int off = 16; off >= 1; off >>= 1) {
      float om = __shfl_xor(m, off, WAVE);
      float os = __shfl_xor(ss, off, WAVE);
      float nm = fmaxf(m, om);
      ss = ss * __expf(m - nm) + os * __expf(om - nm);
      m = nm;
    }
    float lz = m + __logf(ss);
    if (s == r) own_logZ = lz;
  }

  // ---- Phase B: WMMA contraction over K chunks of 32 ----
  const int   myrow = rowBase + s;
  const int   mytgt = target[myrow];
  const float* xrow = adv    + (long long)myrow * C;
  const float* mrow = matrix + (long long)mytgt * C;

  v8f Cacc = {};

  const int nkFull = C >> 5;               // full 32-col chunks, no guards
  for (int kc = 0; kc < nkFull; ++kc) {
    const int k0 = kc << 5;
    __builtin_prefetch(xrow + k0 + 512, 0, 0);   // stream-ahead on adv

    // A: 8-float runs at k0 + hi*16 and k0 + hi*16 + 8 (K=0..15|16..31)
    F8 aLo, aHi;
    const float4* apLo = (const float4*)(mrow + k0 + hi * 16);
    const float4* apHi = (const float4*)(mrow + k0 + hi * 16 + 8);
    aLo.q[0] = apLo[0]; aLo.q[1] = apLo[1];
    aHi.q[0] = apHi[0]; aHi.q[1] = apHi[1];
    // B: 16-float run at k0 + hi*16 (lanes 0-15: K=0..15, lanes 16-31: 16..31)
    F16 bb;
    const float4* bp = (const float4*)(xrow + k0 + hi * 16);
    bb.q[0] = bp[0]; bb.q[1] = bp[1]; bb.q[2] = bp[2]; bb.q[3] = bp[3];

    v16h Av, Bv;
#pragma unroll
    for (int h = 0; h < 8; ++h) {
      Av[h]     = (_Float16)aLo.f[h];
      Av[h + 8] = (_Float16)aHi.f[h];
    }
#pragma unroll
    for (int h = 0; h < 16; ++h)
      Bv[h] = (_Float16)(own_logZ - bb.f[h]);

    Cacc = __builtin_amdgcn_wmma_f32_16x16x32_f16(
        false, Av, false, Bv, (short)0, Cacc, false, false);
  }

  if (C & 31) {                            // branch-free tail chunk (once)
    const int k0 = nkFull << 5;
    v16h Av, Bv;
#pragma unroll
    for (int h = 0; h < 16; ++h) {
      int kA  = h + (hi ? 16 : 0);
      int cA  = k0 + kA;
      int cAc = (cA < C) ? cA : (C - 1);           // clamped, always valid
      float va = mrow[cAc];
      Av[h] = (_Float16)((cA < C) ? va : 0.0f);    // zero-select in regs
      int kB  = h + (hi ? 16 : 0);
      int cB  = k0 + kB;
      int cBc = (cB < C) ? cB : (C - 1);
      float vb = own_logZ - xrow[cBc];
      Bv[h] = (_Float16)((cB < C) ? vb : 0.0f);
    }
    Cacc = __builtin_amdgcn_wmma_f32_16x16x32_f16(
        false, Av, false, Bv, (short)0, Cacc, false, false);
  }

  // Diagonal of D (constant-index, lane-selected; no scratch spill):
  // lane r (0-7) holds diag M=r in element r; lane 24+r holds diag M=r+8
  // in element r.
  float d = 0.0f;
#pragma unroll
  for (int r = 0; r < 8; ++r) {
    float e = Cacc[r];
    d += (lane == r)        ? e : 0.0f;
    d += (lane == (24 + r)) ? e : 0.0f;
  }
  d = wave_sum_f32(d);
  if (lane == 0) atomicAdd(out, d / (float)B);
}

// ---------------------------------------------------------------------------
extern "C" void kernel_launch(void* const* d_in, const int* in_sizes, int n_in,
                              void* d_out, int out_size, void* d_ws, size_t ws_size,
                              hipStream_t stream) {
  const float* clean = (const float*)d_in[0];
  const float* adv   = (const float*)d_in[1];
  const int*   tgt   = (const int*)d_in[2];
  float* out = (float*)d_out;

  const int Bn = in_sizes[2];           // 65536
  const int C  = in_sizes[0] / Bn;      // 1000

  float* acc    = (float*)d_ws;                 // C*C f32
  float* count  = acc + (size_t)C * C;          // C f32
  float* matrix = count + C;                    // C*C f32

  long long nz = ((long long)C * C + C + 3) / 4;
  osl_zero<<<(unsigned)((nz + 255) / 256), 256, 0, stream>>>(acc, count, out, C);

  const int wavesPerBlock = 8;                  // 256 threads = 8 wave32
  osl_stats<<<(Bn + wavesPerBlock - 1) / wavesPerBlock, 256, 0, stream>>>(
      clean, tgt, acc, count, Bn, C);
  osl_matrix<<<(C + wavesPerBlock - 1) / wavesPerBlock, 256, 0, stream>>>(
      acc, count, matrix, C);
  const int nwaves3 = (Bn + 15) / 16;           // 16 rows per wave
  osl_loss<<<(nwaves3 + wavesPerBlock - 1) / wavesPerBlock, 256, 0, stream>>>(
      adv, tgt, matrix, out, Bn, C);
}
